// Model1_13975823581919
// MI455X (gfx1250) — matrix operand order repaired and checked by reading, not compiled, not run on hardware
//
#include <hip/hip_runtime.h>
#include <hip/hip_bf16.h>
#include <cstdint>

typedef __attribute__((ext_vector_type(16))) _Float16 v16h;
typedef __attribute__((ext_vector_type(8)))  _Float16 v8h;
typedef __attribute__((ext_vector_type(8)))  float    v8f;

#define WMMA_F16(A_, B_, C_) \
    __builtin_amdgcn_wmma_f32_16x16x32_f16(false, (A_), false, (B_), (short)0, (C_), false, false)

// ---- model dims (match reference setup_inputs) ----
#define DWORD_ 128      // word embedding dim D
#define HID    256      // GRU hidden H
#define G3     768      // 3*H gate width
#define NSENT  2048     // N sentences
#define TWORD  50       // T words per sentence
#define NDOCS  64       // docs
#define LSENT  32       // sentences per doc
#define PDIM   50       // pos embedding dim P
#define PDOCN  50       // doc pos table rows
#define PSENTN 100      // sent pos table rows

__device__ __forceinline__ float sigmoidf_(float x) { return 1.0f / (1.0f + __expf(-x)); }
// tanh(x) = 2*sigmoid(2x) - 1 : single v_exp_f32 (TRANS pipe, co-executes with WMMA)
__device__ __forceinline__ float tanhf_fast(float x) { return 2.0f / (1.0f + __expf(-2.0f * x)) - 1.0f; }

// ---------------- WMMA fragment loaders (CDNA5 16x16x32 f16 layouts) ----------------
// A matrix 16x32 (MxK), row-major source with leading dim ld (halves):
//   lane r (0-15): halves 0..7 = K[kb..kb+7], halves 8..15 = K[16+kb..16+kb+7], kb = 8*(lane>=16)
__device__ __forceinline__ v16h ld_fragA(const _Float16* base, size_t ld, int kk, int lane) {
    const int r  = lane & 15;
    const int kb = (lane >> 4) << 3;
    const _Float16* p = base + (size_t)r * ld + (size_t)(kk << 5) + kb;
    v8h lo = *(const v8h*)(p);
    v8h hi = *(const v8h*)(p + 16);
    v16h f;
#pragma unroll
    for (int i = 0; i < 8; ++i) { f[i] = lo[i]; f[i + 8] = hi[i]; }
    return f;
}

__device__ __forceinline__ v16h ld_fragA_guard(const _Float16* base, size_t ld, int kk,
                                               int lane, int rowsValid) {
    const int r = lane & 15;
    v16h f;
    if (r < rowsValid) {
        const int kb = (lane >> 4) << 3;
        const _Float16* p = base + (size_t)r * ld + (size_t)(kk << 5) + kb;
        v8h lo = *(const v8h*)(p);
        v8h hi = *(const v8h*)(p + 16);
#pragma unroll
        for (int i = 0; i < 8; ++i) { f[i] = lo[i]; f[i + 8] = hi[i]; }
    } else {
#pragma unroll
        for (int i = 0; i < 16; ++i) f[i] = (_Float16)0.0f;
    }
    return f;
}

// B matrix 32x16 (KxN), source is [N,K] row-major (i.e. the weight matrix [gate, k]):
//   lane n (0-15): halves 0..7 = K[kb..kb+7], halves 8..15 = K[16+kb..16+kb+7], kb = 8*(lane>=16)
__device__ __forceinline__ v16h ld_fragB(const _Float16* base, size_t ld, int kk, int lane) {
    const int nn = lane & 15;
    const int kb = (lane >> 4) << 3;
    const _Float16* p = base + (size_t)nn * ld + (size_t)(kk << 5) + kb;
    v8h lo = *(const v8h*)(p);
    v8h hi = *(const v8h*)(p + 16);
    v16h f;
#pragma unroll
    for (int i = 0; i < 8; ++i) { f[i] = lo[i]; f[i + 8] = hi[i]; }
    return f;
}

// ---------------- small helper kernels ----------------
__global__ void k_f32_to_f16(const float* __restrict__ s, _Float16* __restrict__ d, int n) {
    int i = blockIdx.x * blockDim.x + threadIdx.x;
    if (i < n) d[i] = (_Float16)s[i];
}

__global__ void k_gather_embed(const int* __restrict__ x, const float* __restrict__ embed,
                               _Float16* __restrict__ e16, int ntok) {
    int i = blockIdx.x * blockDim.x + threadIdx.x;    // over ntok*128 elements
    if (i < ntok * DWORD_) {
        int tok = i >> 7;
        int dd  = i & 127;
        int id  = x[tok];
        e16[i] = (_Float16)embed[(size_t)id * DWORD_ + dd];
    }
}

__global__ void k_lens(const int* __restrict__ x, int* __restrict__ lens, int B, int T) {
    int nidx = blockIdx.x * blockDim.x + threadIdx.x;
    if (nidx < B) {
        int c = 0;
        for (int t = 0; t < T; ++t) c += (x[nidx * T + t] != 0) ? 1 : 0;
        lens[nidx] = c;
    }
}

// ---------------- fused BiGRU direction scan with WMMA ----------------
// One block = 16 batch rows.  8 waves * 6 tiles cover the 768 gate columns with
// tile index ti = wave + 8*c, so columns j, j+256, j+512 live in the same wave at
// c, c+2, c+4 -> gate math + h state + masked max pooling stay in registers.
// Per step: acc = ein[t] @ Wi^T (+ separate accx for the n-gate) then += h @ Wh^T.
// DIN: compile-time input width (128 word-level, 512 sentence/doc level).
// FULL: block is a full 16-row tile (no lane guards in the hot path).
template <int DIN, bool FULL>
__global__ __launch_bounds__(256)
void k_gru_scan(const _Float16* __restrict__ ein,
                const _Float16* __restrict__ Wi16, const float* __restrict__ bi,
                const _Float16* __restrict__ Wh16, const float* __restrict__ bh,
                const int* __restrict__ lens,
                int B, int T, int rev,
                float* __restrict__ out_pool, _Float16* __restrict__ out_pool16, int dirOff)
{
    __shared__ __align__(16) _Float16 sh_h16[16 * 264];   // h state tile, padded ld=264

    const int tid  = threadIdx.x;
    const int wave = tid >> 5;
    const int lane = tid & 31;
    const int nn   = lane & 15;
    const int mb   = (lane >> 4) << 3;
    const int row0 = blockIdx.x * 16;
    const int rowsValid = (B - row0) < 16 ? (B - row0) : 16;

    for (int i = tid; i < 16 * 264; i += 256) sh_h16[i] = (_Float16)0.0f;

    float hreg[2][8], vmax[2][8];
    float biR[2], biZ[2], biN[2], bhR[2], bhZ[2], bhN[2];
    int   lenr[8];
#pragma unroll
    for (int c = 0; c < 2; ++c) {
        const int col = (wave + 8 * c) * 16 + nn;         // r-gate column j
        biR[c] = bi[col];        bhR[c] = bh[col];
        biZ[c] = bi[col + 256];  bhZ[c] = bh[col + 256];
        biN[c] = bi[col + 512];  bhN[c] = bh[col + 512];
#pragma unroll
        for (int v = 0; v < 8; ++v) { hreg[c][v] = 0.0f; vmax[c][v] = -3.0e38f; }
    }
#pragma unroll
    for (int v = 0; v < 8; ++v) {
        const int row = row0 + mb + v;
        lenr[v] = (row < B) ? (lens ? lens[row] : T) : 0;
    }

    const size_t ldA = (size_t)T * DIN;
    constexpr int kIn = DIN >> 5;

    for (int s = 0; s < T; ++s) {
        const int t = rev ? (T - 1 - s) : s;
        __syncthreads();                                  // h16 from previous step visible
        // Opaque memory clobber: stops LICM from hoisting the (loop-invariant)
        // weight-fragment loads out of the time loop -- they must stream from
        // L2 each step instead of being preloaded into registers and spilled.
        asm volatile("" ::: "memory");

        v8f acc[6], accx[2];
#pragma unroll
        for (int c = 0; c < 6; ++c)
#pragma unroll
            for (int v = 0; v < 8; ++v) acc[c][v] = 0.0f;
#pragma unroll
        for (int c = 0; c < 2; ++c)
#pragma unroll
            for (int v = 0; v < 8; ++v) accx[c][v] = 0.0f;

        // ---- input GEMM: x_t @ Wi^T (K = DIN) ----
        const _Float16* abase = ein + ((size_t)row0 * T + t) * DIN;
#pragma unroll 2
        for (int kk = 0; kk < kIn; ++kk) {
            v16h a = FULL ? ld_fragA(abase, ldA, kk, lane)
                          : ld_fragA_guard(abase, ldA, kk, lane, rowsValid);
#pragma unroll
            for (int c = 0; c < 6; ++c) {
                const int col0 = (wave + 8 * c) * 16;
                v16h b = ld_fragB(Wi16 + (size_t)col0 * DIN, DIN, kk, lane);
                if (c >= 4) accx[c - 4] = WMMA_F16(a, b, accx[c - 4]);   // n-gate x part
                else        acc[c]      = WMMA_F16(a, b, acc[c]);
            }
        }
        // ---- recurrent GEMM: h @ Wh^T (K = 256) ----
#pragma unroll 2
        for (int kk = 0; kk < 8; ++kk) {
            v16h a = ld_fragA(sh_h16, 264, kk, lane);
#pragma unroll
            for (int c = 0; c < 6; ++c) {
                const int col0 = (wave + 8 * c) * 16;
                v16h b = ld_fragB(Wh16 + (size_t)col0 * HID, HID, kk, lane);
                acc[c] = WMMA_F16(a, b, acc[c]);
            }
        }
        __syncthreads();                                  // all waves done reading h16

        // ---- gates (all in registers) + write new h16 ----
#pragma unroll
        for (int c = 0; c < 2; ++c) {
            const int j = (wave + 8 * c) * 16 + nn;
#pragma unroll
            for (int v = 0; v < 8; ++v) {
                float r    = sigmoidf_(acc[c][v]     + biR[c] + bhR[c]);
                float z    = sigmoidf_(acc[c + 2][v] + biZ[c] + bhZ[c]);
                float hn   = acc[c + 4][v] + bhN[c];
                float xn   = accx[c][v]    + biN[c];
                float nnew = tanhf_fast(xn + r * hn);
                float hnew = (1.0f - z) * nnew + z * hreg[c][v];
                hreg[c][v] = hnew;
                if (t < lenr[v]) vmax[c][v] = fmaxf(vmax[c][v], hnew);
                sh_h16[(mb + v) * 264 + j] = (_Float16)hnew;
            }
        }
    }

    // ---- masked max-pool outputs (0 when len==0) ----
#pragma unroll
    for (int c = 0; c < 2; ++c) {
        const int j = (wave + 8 * c) * 16 + nn;
#pragma unroll
        for (int v = 0; v < 8; ++v) {
            const int row = row0 + mb + v;
            if (row < B) {
                float val = (lenr[v] > 0) ? vmax[c][v] : 0.0f;
                out_pool[(size_t)row * (2 * HID) + dirOff + j] = val;
                if (out_pool16)
                    out_pool16[(size_t)row * (2 * HID) + dirOff + j] = (_Float16)val;
            }
        }
    }
}

// ---------------- context transform: u[nd,:] = W_sal @ concat(blog, doc_vec[nd]) ----------------
__global__ void k_context_u(const float* __restrict__ W_sal, const float* __restrict__ blog,
                            const float* __restrict__ doc_vec, float* __restrict__ u) {
    int d = blockIdx.x;   // doc
    for (int o = threadIdx.x; o < 2 * HID; o += blockDim.x) {
        float sacc = 0.0f;
        const float* wrow = W_sal + (size_t)o * (4 * HID);
        for (int e = 0; e < 2 * HID; ++e)  sacc += wrow[e] * blog[e];
        for (int e = 0; e < 2 * HID; ++e)  sacc += wrow[2 * HID + e] * doc_vec[(size_t)d * (2 * HID) + e];
        u[(size_t)d * (2 * HID) + o] = sacc;
    }
}

// ---------------- final scores ----------------
__global__ void k_score(const float* __restrict__ sent_vec, const float* __restrict__ u,
                        const float* __restrict__ w_content,
                        const float* __restrict__ doc_pos, const float* __restrict__ sent_pos,
                        const float* __restrict__ w_dpos, const float* __restrict__ w_spos,
                        const float* __restrict__ sent_bias, const int* __restrict__ doc_lens,
                        float* __restrict__ out) {
    int i = blockIdx.x * blockDim.x + threadIdx.x;
    if (i >= NSENT) return;
    int d = i / LSENT, l = i % LSENT;
    const float* sv = sent_vec + (size_t)i * (2 * HID);
    const float* uu = u + (size_t)d * (2 * HID);
    float content = 0.0f, sal = 0.0f;
    for (int e = 0; e < 2 * HID; ++e) { content += sv[e] * w_content[e]; sal += sv[e] * uu[e]; }
    int di = (d * PDOCN) / NDOCS;
    float dsc = 0.0f;
    for (int p = 0; p < PDIM; ++p) dsc += doc_pos[di * PDIM + p] * w_dpos[p];
    int dl = doc_lens[d];
    int safe = dl >= 1 ? dl : 1;
    int si = (l * PSENTN) / safe;
    if (si > PSENTN - 1) si = PSENTN - 1;
    float ssc = 0.0f;
    for (int p = 0; p < PDIM; ++p) ssc += sent_pos[si * PDIM + p] * w_spos[p];
    out[i] = content + sal + dsc + ssc + sent_bias[0];
}

extern "C" void kernel_launch(void* const* d_in, const int* in_sizes, int n_in,
                              void* d_out, int out_size, void* d_ws, size_t ws_size,
                              hipStream_t stream) {
    (void)in_sizes; (void)n_in; (void)out_size; (void)ws_size;

    const int*   x        = (const int*)d_in[0];
    const int*   doc_lens = (const int*)d_in[1];
    const float* embed    = (const float*)d_in[2];
    const float* wWi_f = (const float*)d_in[3];
    const float* wWh_f = (const float*)d_in[4];
    const float* wbi_f = (const float*)d_in[5];
    const float* wbh_f = (const float*)d_in[6];
    const float* wWi_b = (const float*)d_in[7];
    const float* wWh_b = (const float*)d_in[8];
    const float* wbi_b = (const float*)d_in[9];
    const float* wbh_b = (const float*)d_in[10];
    const float* sWi_f = (const float*)d_in[11];
    const float* sWh_f = (const float*)d_in[12];
    const float* sbi_f = (const float*)d_in[13];
    const float* sbh_f = (const float*)d_in[14];
    const float* sWi_b = (const float*)d_in[15];
    const float* sWh_b = (const float*)d_in[16];
    const float* sbi_b = (const float*)d_in[17];
    const float* sbh_b = (const float*)d_in[18];
    const float* dWi_f = (const float*)d_in[19];
    const float* dWh_f = (const float*)d_in[20];
    const float* dbi_f = (const float*)d_in[21];
    const float* dbh_f = (const float*)d_in[22];
    const float* dWi_b = (const float*)d_in[23];
    const float* dWh_b = (const float*)d_in[24];
    const float* dbi_b = (const float*)d_in[25];
    const float* dbh_b = (const float*)d_in[26];
    const float* doc_pos   = (const float*)d_in[27];
    const float* sent_pos  = (const float*)d_in[28];
    const float* w_content = (const float*)d_in[29];
    const float* W_sal     = (const float*)d_in[30];
    const float* w_dpos    = (const float*)d_in[31];
    const float* w_spos    = (const float*)d_in[32];
    const float* sent_bias = (const float*)d_in[33];

    // ---- workspace layout (256B aligned) ----
    char* ws = (char*)d_ws;
    size_t off = 0;
    auto alloc = [&](size_t bytes) -> char* {
        char* p = ws + off;
        off += (bytes + 255) & ~(size_t)255;
        return p;
    };
    _Float16* e16   = (_Float16*)alloc((size_t)NSENT * TWORD * DWORD_ * 2);
    _Float16* wWiF  = (_Float16*)alloc((size_t)G3 * DWORD_ * 2);
    _Float16* wWiB  = (_Float16*)alloc((size_t)G3 * DWORD_ * 2);
    _Float16* wWhF  = (_Float16*)alloc((size_t)G3 * HID * 2);
    _Float16* wWhB  = (_Float16*)alloc((size_t)G3 * HID * 2);
    _Float16* sWiF  = (_Float16*)alloc((size_t)G3 * 2 * HID * 2);
    _Float16* sWiB  = (_Float16*)alloc((size_t)G3 * 2 * HID * 2);
    _Float16* sWhF  = (_Float16*)alloc((size_t)G3 * HID * 2);
    _Float16* sWhB  = (_Float16*)alloc((size_t)G3 * HID * 2);
    _Float16* dWiF  = (_Float16*)alloc((size_t)G3 * 2 * HID * 2);
    _Float16* dWiB  = (_Float16*)alloc((size_t)G3 * 2 * HID * 2);
    _Float16* dWhF  = (_Float16*)alloc((size_t)G3 * HID * 2);
    _Float16* dWhB  = (_Float16*)alloc((size_t)G3 * HID * 2);
    int*      sent_lens = (int*)alloc((size_t)NSENT * 4);
    float*    sent_vec  = (float*)alloc((size_t)NSENT * 2 * HID * 4);
    _Float16* sent_v16  = (_Float16*)alloc((size_t)NSENT * 2 * HID * 2);
    float*    doc_vec   = (float*)alloc((size_t)NDOCS * 2 * HID * 4);
    _Float16* doc_v16   = (_Float16*)alloc((size_t)NDOCS * 2 * HID * 2);
    float*    blog      = (float*)alloc((size_t)2 * HID * 4);
    float*    u         = (float*)alloc((size_t)NDOCS * 2 * HID * 4);

    auto conv = [&](const float* s, _Float16* d, int ncount) {
        k_f32_to_f16<<<(ncount + 255) / 256, 256, 0, stream>>>(s, d, ncount);
    };

    // ---- weight conversion to f16 ----
    conv(wWi_f, wWiF, G3 * DWORD_);   conv(wWi_b, wWiB, G3 * DWORD_);
    conv(wWh_f, wWhF, G3 * HID);      conv(wWh_b, wWhB, G3 * HID);
    conv(sWi_f, sWiF, G3 * 2 * HID);  conv(sWi_b, sWiB, G3 * 2 * HID);
    conv(sWh_f, sWhF, G3 * HID);      conv(sWh_b, sWhB, G3 * HID);
    conv(dWi_f, dWiF, G3 * 2 * HID);  conv(dWi_b, dWiB, G3 * 2 * HID);
    conv(dWh_f, dWhF, G3 * HID);      conv(dWh_b, dWhB, G3 * HID);

    // ---- embedding gather (f16) + sentence lengths ----
    {
        int total = NSENT * TWORD * DWORD_;
        k_gather_embed<<<(total + 255) / 256, 256, 0, stream>>>(x, embed, e16, NSENT * TWORD);
        k_lens<<<(NSENT + 255) / 256, 256, 0, stream>>>(x, sent_lens, NSENT, TWORD);
    }

    // ---- word-level BiGRU -> sent_vec [2048, 512] (B%16==0 -> FULL path) ----
    {
        int grid = NSENT / 16;
        k_gru_scan<DWORD_, true><<<grid, 256, 0, stream>>>(e16, wWiF, wbi_f, wWhF, wbh_f,
                                                           sent_lens, NSENT, TWORD, 0,
                                                           sent_vec, sent_v16, 0);
        k_gru_scan<DWORD_, true><<<grid, 256, 0, stream>>>(e16, wWiB, wbi_b, wWhB, wbh_b,
                                                           sent_lens, NSENT, TWORD, 1,
                                                           sent_vec, sent_v16, HID);
    }

    // ---- sentence-level BiGRU -> doc_vec [64, 512] (B%16==0 -> FULL path) ----
    {
        int grid = NDOCS / 16;
        k_gru_scan<2 * HID, true><<<grid, 256, 0, stream>>>(sent_v16, sWiF, sbi_f, sWhF, sbh_f,
                                                            doc_lens, NDOCS, LSENT, 0,
                                                            doc_vec, doc_v16, 0);
        k_gru_scan<2 * HID, true><<<grid, 256, 0, stream>>>(sent_v16, sWiB, sbi_b, sWhB, sbh_b,
                                                            doc_lens, NDOCS, LSENT, 1,
                                                            doc_vec, doc_v16, HID);
    }

    // ---- doc-level BiGRU over 64 "timesteps" (B=1 -> guarded path); pooled max == blog_vec ----
    {
        k_gru_scan<2 * HID, false><<<1, 256, 0, stream>>>(doc_v16, dWiF, dbi_f, dWhF, dbh_f,
                                                          nullptr, 1, NDOCS, 0, blog, nullptr, 0);
        k_gru_scan<2 * HID, false><<<1, 256, 0, stream>>>(doc_v16, dWiB, dbi_b, dWhB, dbh_b,
                                                          nullptr, 1, NDOCS, 1, blog, nullptr, HID);
    }

    // ---- context transform + scores ----
    k_context_u<<<NDOCS, 256, 0, stream>>>(W_sal, blog, doc_vec, u);
    k_score<<<(NSENT + 255) / 256, 256, 0, stream>>>(sent_vec, u, w_content, doc_pos, sent_pos,
                                                     w_dpos, w_spos, sent_bias, doc_lens,
                                                     (float*)d_out);
}
